// _System_general_static_21406117003442
// MI455X (gfx1250) — compile-verified, matrix operand order checked by reading, not run on hardware
//
#include <hip/hip_runtime.h>

typedef __attribute__((ext_vector_type(16))) _Float16 v16h;
typedef __attribute__((ext_vector_type(8)))  float    v8f;
typedef __attribute__((ext_vector_type(2)))  _Float16 h2;
typedef __attribute__((ext_vector_type(2)))  float    f2;   // native vector for nontemporal loads

#define WAVES_PER_BLOCK 8
#define TILE_ROWS 16
#define ROWS_PER_BLOCK (WAVES_PER_BLOCK * TILE_ROWS)

// Unconditional probe: CDNA5 lists TANH as a native VALU transcendental.
__device__ __forceinline__ float fast_tanh(float x) {
  float y;
  asm("v_tanh_f32 %0, %1" : "=v"(y) : "v"(x));
  return y;
}

__global__ __launch_bounds__(256) void ren_forward_kernel(
    const float* __restrict__ u,
    const float* __restrict__ D11,
    const float* __restrict__ D12,
    const float* __restrict__ D21,
    const float* __restrict__ D22,
    const float* __restrict__ bv,
    const float* __restrict__ by,
    float* __restrict__ out)
{
  // ---- shared staging ----
  __shared__ _Float16 sD12[64 * 64];            // row-major [q][k]  (B operand for du)
  __shared__ _Float16 sD21[64 * 64];            // row-major [y][k]  (B operand for w@D21^T)
  __shared__ _Float16 sD22[64 * 64];            // row-major [y][k]  (B operand for u@D22^T)
  __shared__ __align__(16) float sLt[64 * 64];  // [(j*16 + n%16)*4 + n/16], strict-lower masked
  __shared__ float sbv[64];
  __shared__ float sby[64];
  // One per-wave 16x64 f16 scratch tile, time-shared: holds u (f16) during the
  // recurrence, then is overwritten with w for the output GEMM's A rebuild.
  __shared__ __align__(4) _Float16 sUW[WAVES_PER_BLOCK][16 * 64];

  const int tid = threadIdx.x;
  for (int e = tid; e < 64 * 64; e += 256) {
    sD12[e] = (_Float16)D12[e];
    sD21[e] = (_Float16)D21[e];
    sD22[e] = (_Float16)D22[e];
    const int j = e >> 6;          // source column of recurrence
    const int n = e & 63;          // destination row
    // L = tril(D11, -1): zero on/above diagonal so finalized columns get +0
    sLt[(j * 16 + (n & 15)) * 4 + (n >> 4)] = (n > j) ? D11[n * 64 + j] : 0.0f;
  }
  if (tid < 64) { sbv[tid] = bv[tid]; sby[tid] = by[tid]; }
  __syncthreads();

  const int lane  = tid & 31;
  const int wave  = tid >> 5;
  const int half  = lane >> 4;      // 0: lanes 0-15, 1: lanes 16-31
  const int colid = lane & 15;
  const long rowBase = (long)blockIdx.x * ROWS_PER_BLOCK + (long)wave * TILE_ROWS;
  const long mA = rowBase + colid;  // this lane's A-operand row (M = lane%16)
  _Float16* __restrict__ myuw = sUW[wave];

  // ---- load u tile (16x64 f32) into WMMA f16 A-layout regs AND stash f16 copy in LDS ----
  // 16-bit A layout: lane l, VGPR r holds K0,K0+1 with K0 = 2*(r&3) + 16*(r>>2) + 8*half
  v16h uA[2];
#pragma unroll
  for (int s = 0; s < 2; ++s) {
#pragma unroll
    for (int r = 0; r < 8; ++r) {
      const int K0 = 32 * s + 2 * (r & 3) + ((r >> 2) << 4) + (half << 3);
      const f2 f = __builtin_nontemporal_load((const f2*)(u + mA * 64 + K0));
      const _Float16 h0 = (_Float16)f[0], h1 = (_Float16)f[1];
      uA[s][2 * r]     = h0;
      uA[s][2 * r + 1] = h1;
      h2 p; p[0] = h0; p[1] = h1;
      *(h2*)(myuw + colid * 64 + K0) = p;   // lanes colid / colid+16 cover all 64 k
    }
  }

  // Build a B operand (K=32 slice s, output-column tile nt) from a row-major
  // LDS f16 matrix M[n][k]; B[k][n] = M[nt*16 + n][k]. Mirrored lane layout: N = lane%16.
  auto buildB = [&](const _Float16* __restrict__ M, int nt, int s) -> v16h {
    v16h b;
#pragma unroll
    for (int r = 0; r < 8; ++r) {
      const int K0 = 32 * s + 2 * (r & 3) + ((r >> 2) << 4) + (half << 3);
      const h2 v = *(const h2*)(M + (nt * 16 + colid) * 64 + K0);
      b[2 * r]     = v[0];
      b[2 * r + 1] = v[1];
    }
    return b;
  };

  // ---- du = u @ D12^T + bv  ->  acc[4][8] in WMMA C-layout (row = r + 8*half, col = 16*b + colid)
  float acc[4][8];
#pragma unroll
  for (int b = 0; b < 4; ++b) {
    v8f c = {};
#pragma unroll
    for (int s = 0; s < 2; ++s) {
      const v16h bb = buildB(sD12, b, s);
      c = __builtin_amdgcn_wmma_f32_16x16x32_f16(false, uA[s], false, bb,
                                                 (short)0, c, false, false);
    }
    const float biasv = sbv[b * 16 + colid];
#pragma unroll
    for (int r = 0; r < 8; ++r) acc[b][r] = c[r] + biasv;
  }
  // uA dies here -> recurrence runs with ~45 live VGPRs.

  // ---- triangular tanh recurrence over the 64 columns ----
  // Step j: every lane fetches column j's accumulator (lanes j%16 / j%16+16 of
  // block j/16) via bpermute, applies tanh -> w_j, FMAs L[.,j] into all 4 column
  // blocks (upper triangle pre-zeroed), owners overwrite acc with w_j.
#pragma unroll
  for (int bs = 0; bs < 4; ++bs) {
    for (int jj = 0; jj < 16; ++jj) {
      const int j = bs * 16 + jj;
      float wj[8];
#pragma unroll
      for (int r = 0; r < 8; ++r) {
        const float t = __shfl(acc[bs][r], (lane & 16) | jj, 32);
        wj[r] = fast_tanh(t);
      }
      const float4 Lc = *(const float4*)(sLt + (j * 16 + colid) * 4);
#pragma unroll
      for (int r = 0; r < 8; ++r) {
        acc[0][r] = fmaf(Lc.x, wj[r], acc[0][r]);
        acc[1][r] = fmaf(Lc.y, wj[r], acc[1][r]);
        acc[2][r] = fmaf(Lc.z, wj[r], acc[2][r]);
        acc[3][r] = fmaf(Lc.w, wj[r], acc[3][r]);
      }
      if (colid == jj) {
#pragma unroll
        for (int r = 0; r < 8; ++r) acc[bs][r] = wj[r];
      }
    }
  }

  // ---- rebuild uA from LDS (reads must precede the w overwrite; DS is in-order per wave) ----
#pragma unroll
  for (int s = 0; s < 2; ++s) {
#pragma unroll
    for (int r = 0; r < 8; ++r) {
      const int K0 = 32 * s + 2 * (r & 3) + ((r >> 2) << 4) + (half << 3);
      const h2 v = *(const h2*)(myuw + colid * 64 + K0);
      uA[s][2 * r]     = v[0];
      uA[s][2 * r + 1] = v[1];
    }
  }
  asm volatile("" ::: "memory");  // keep u reloads ordered before the w overwrite

  // ---- overwrite the scratch tile with w (C-layout -> [m][k] f16) ----
#pragma unroll
  for (int b = 0; b < 4; ++b)
#pragma unroll
    for (int r = 0; r < 8; ++r)
      myuw[(r + 8 * half) * 64 + b * 16 + colid] = (_Float16)acc[b][r];

  asm volatile("s_wait_dscnt 0" ::: "memory");  // intra-wave DS store->load ordering

  // ---- rebuild w in A-layout ----
  v16h wA[2];
#pragma unroll
  for (int s = 0; s < 2; ++s) {
#pragma unroll
    for (int r = 0; r < 8; ++r) {
      const int K0 = 32 * s + 2 * (r & 3) + ((r >> 2) << 4) + (half << 3);
      const h2 v = *(const h2*)(myuw + colid * 64 + K0);
      wA[s][2 * r]     = v[0];
      wA[s][2 * r + 1] = v[1];
    }
  }

  // ---- yt = w @ D21^T + u @ D22^T + by, streamed out per 16-column tile ----
#pragma unroll
  for (int nt = 0; nt < 4; ++nt) {
    v8f c = {};
#pragma unroll
    for (int s = 0; s < 2; ++s) {
      const v16h b21 = buildB(sD21, nt, s);
      c = __builtin_amdgcn_wmma_f32_16x16x32_f16(false, wA[s], false, b21,
                                                 (short)0, c, false, false);
    }
#pragma unroll
    for (int s = 0; s < 2; ++s) {
      const v16h b22 = buildB(sD22, nt, s);
      c = __builtin_amdgcn_wmma_f32_16x16x32_f16(false, uA[s], false, b22,
                                                 (short)0, c, false, false);
    }
    const float biasv = sby[nt * 16 + colid];
#pragma unroll
    for (int r = 0; r < 8; ++r) {
      const long m = rowBase + r + 8 * half;
      __builtin_nontemporal_store(c[r] + biasv, out + m * 64 + nt * 16 + colid);
    }
  }
}

extern "C" void kernel_launch(void* const* d_in, const int* in_sizes, int n_in,
                              void* d_out, int out_size, void* d_ws, size_t ws_size,
                              hipStream_t stream) {
  // setup_inputs order: t(0), xi(1), u(2), D11(3), D12(4), D21(5), D22(6), bv(7), by(8)
  const float* u   = (const float*)d_in[2];
  const float* D11 = (const float*)d_in[3];
  const float* D12 = (const float*)d_in[4];
  const float* D21 = (const float*)d_in[5];
  const float* D22 = (const float*)d_in[6];
  const float* bv  = (const float*)d_in[7];
  const float* by  = (const float*)d_in[8];
  float* out = (float*)d_out;

  const int batch  = in_sizes[2] / 64;           // 262144
  const int blocks = batch / ROWS_PER_BLOCK;     // 2048
  ren_forward_kernel<<<dim3(blocks), dim3(256), 0, stream>>>(
      u, D11, D12, D21, D22, bv, by, out);
}
